// Seq2Seq_82171314307466
// MI455X (gfx1250) — compile-verified, matrix operand order checked by reading, not run on hardware
//
#include <hip/hip_runtime.h>
#include <hip/hip_bf16.h>

// ---------------------------------------------------------------------------
// Seq2Seq (bi-LSTM encoder + LSTM decoder + big FC) for MI455X / gfx1250.
// All matrix math via v_wmma_f32_16x16x32_f16 (wave32 WMMA), f32 accumulate.
// FC kernel: 2x5 register-blocked wave tiles (10 accumulators) for ~3x less
// L2 traffic; gate kernel: 2 M-tiles per wave so Whh is streamed once.
// ---------------------------------------------------------------------------

typedef _Float16 v16h __attribute__((ext_vector_type(16)));
typedef _Float16 v8h  __attribute__((ext_vector_type(8)));
typedef float    v8f  __attribute__((ext_vector_type(8)));

#define CB  32          // batch
#define CS  96          // src len
#define CT  96          // tgt len
#define CE  200         // embed dim
#define CKP 224         // embed dim padded to multiple of 32
#define CH  512         // hidden
#define CG  2048        // 4*H
#define CV  30000       // vocab
#define CTm1 95         // decoder steps

static __device__ __forceinline__ float sigf(float x) {
    return 1.0f / (1.0f + __expf(-x));
}

// Load one 16x32 f16 A/B fragment per the CDNA5 16-bit A layout:
// lane%16 = row, lane/16 selects K-chunk; elements 0..7 at K=half*8,
// elements 8..15 at K=16+half*8. p must point at rowBase + k0 + half*8.
static __device__ __forceinline__ v16h load_frag16(const _Float16* p) {
    v8h lo = *reinterpret_cast<const v8h*>(p);
    v8h hi = *reinterpret_cast<const v8h*>(p + 16);
    v16h r;
#pragma unroll
    for (int i = 0; i < 8; ++i) { r[i] = lo[i]; r[i + 8] = hi[i]; }
    return r;
}

static __device__ __forceinline__ v8f wmma_f16(v16h a, v16h b, v8f c) {
    return __builtin_amdgcn_wmma_f32_16x16x32_f16(
        false, a, false, b, (short)0, c, false, false);
}

// ---------------------------------------------------------------------------
// Utility kernels
// ---------------------------------------------------------------------------

__global__ void zero_u32_kernel(unsigned int* p, long n) {
    long i = (long)blockIdx.x * blockDim.x + threadIdx.x;
    if (i < n) p[i] = 0u;
}

// out[:, 0, :] = 0
__global__ void zero_t0_kernel(float* out) {
    long i = (long)blockIdx.x * blockDim.x + threadIdx.x;
    if (i >= (long)CB * CV) return;
    long b = i / CV, v = i % CV;
    out[b * (long)CT * CV + v] = 0.0f;
}

// Gather encoder embeddings (forward + time-reversed), fp32 -> f16, pad K.
__global__ void enc_gather_kernel(const int* __restrict__ tok,
                                  const float* __restrict__ emb,
                                  _Float16* __restrict__ Xf,
                                  _Float16* __restrict__ Xb) {
    long i = (long)blockIdx.x * blockDim.x + threadIdx.x;
    if (i >= (long)CB * CS * CKP) return;
    int b = (int)(i / (CS * CKP));
    int r = (int)(i % (CS * CKP));
    int s = r / CKP, e = r % CKP;
    _Float16 vf = (_Float16)0.0f, vb = (_Float16)0.0f;
    if (e < CE) {
        int tf = tok[b * CS + s];
        int tb = tok[b * CS + (CS - 1 - s)];
        vf = (_Float16)emb[(long)tf * CE + e];
        vb = (_Float16)emb[(long)tb * CE + e];
    }
    Xf[i] = vf;
    Xb[i] = vb;
}

// Gather decoder embeddings for dec_in = Linear_formula[:, :T-1].
__global__ void dec_gather_kernel(const int* __restrict__ tok,
                                  const float* __restrict__ emb,
                                  _Float16* __restrict__ Xd) {
    long i = (long)blockIdx.x * blockDim.x + threadIdx.x;
    if (i >= (long)CB * CTm1 * CKP) return;
    int b = (int)(i / (CTm1 * CKP));
    int r = (int)(i % (CTm1 * CKP));
    int t = r / CKP, e = r % CKP;
    _Float16 v = (_Float16)0.0f;
    if (e < CE) {
        int tk = tok[b * CT + t];
        v = (_Float16)emb[(long)tk * CE + e];
    }
    Xd[i] = v;
}

// fp32 (rows x K) -> f16 (rows x Kp), zero-padded columns K..Kp-1.
__global__ void cvt_pad_kernel(const float* __restrict__ src,
                               _Float16* __restrict__ dst,
                               long rows, int K, int Kp) {
    long i = (long)blockIdx.x * blockDim.x + threadIdx.x;
    if (i >= rows * Kp) return;
    long r = i / Kp;
    int  k = (int)(i % Kp);
    dst[i] = (k < K) ? (_Float16)src[r * K + k] : (_Float16)0.0f;
}

// Build [h_f | h_b] (f16) and [c_f | c_b] (f16) for the h0/c0 GEMMs.
__global__ void concat_hc_kernel(const _Float16* __restrict__ hf,
                                 const _Float16* __restrict__ hb,
                                 const float* __restrict__ cf,
                                 const float* __restrict__ cb,
                                 _Float16* __restrict__ hcat,
                                 _Float16* __restrict__ ccat) {
    int i = blockIdx.x * blockDim.x + threadIdx.x;
    if (i >= CB * 2 * CH) return;
    int b = i / (2 * CH), j = i % (2 * CH);
    if (j < CH) {
        hcat[i] = hf[b * CH + j];
        ccat[i] = (_Float16)cf[b * CH + j];
    } else {
        hcat[i] = hb[b * CH + (j - CH)];
        ccat[i] = (_Float16)cb[b * CH + (j - CH)];
    }
}

// ---------------------------------------------------------------------------
// WMMA kernels
// ---------------------------------------------------------------------------

// Gate pre-activation for one LSTM step:
//   G(32 x 2048) = Xt(32 x 224) @ Wih(2048 x 224)^T
//               + h(32 x 512)  @ Whh(2048 x 512)^T + bias
// 128 waves, one N-tile each, BOTH M-tiles per wave (weight rows read once).
__global__ __launch_bounds__(256)
void lstm_gates_kernel(const _Float16* __restrict__ Xt, int ldx,
                       const _Float16* __restrict__ Wih,
                       const _Float16* __restrict__ h16,
                       const _Float16* __restrict__ Whh,
                       const float* __restrict__ bias,
                       float* __restrict__ G) {
    int ntile = blockIdx.x * (blockDim.x >> 5) + (threadIdx.x >> 5);  // 0..127
    int lane  = threadIdx.x & 31;
    int row   = lane & 15;
    int half  = lane >> 4;

    v8f acc0 = {};
    v8f acc1 = {};

    // x @ Wih^T  (K = 224)
    {
        const _Float16* a0 = Xt  + (long)row        * ldx + half * 8;
        const _Float16* a1 = Xt  + (long)(16 + row) * ldx + half * 8;
        const _Float16* bw = Wih + (long)(ntile * 16 + row) * CKP + half * 8;
#pragma unroll
        for (int k0 = 0; k0 < CKP; k0 += 32) {
            v16h b  = load_frag16(bw + k0);
            v16h af0 = load_frag16(a0 + k0);
            v16h af1 = load_frag16(a1 + k0);
            acc0 = wmma_f16(af0, b, acc0);
            acc1 = wmma_f16(af1, b, acc1);
        }
    }
    // h @ Whh^T  (K = 512)
    {
        const _Float16* a0 = h16 + (long)row        * CH + half * 8;
        const _Float16* a1 = h16 + (long)(16 + row) * CH + half * 8;
        const _Float16* bw = Whh + (long)(ntile * 16 + row) * CH + half * 8;
#pragma unroll 4
        for (int k0 = 0; k0 < CH; k0 += 32) {
            v16h b  = load_frag16(bw + k0);
            v16h af0 = load_frag16(a0 + k0);
            v16h af1 = load_frag16(a1 + k0);
            acc0 = wmma_f16(af0, b, acc0);
            acc1 = wmma_f16(af1, b, acc1);
        }
    }

    int   col = ntile * 16 + (lane & 15);
    float bb  = bias[col];
#pragma unroll
    for (int r = 0; r < 8; ++r) {
        int mrow = r + half * 8;
        G[(long)mrow * CG + col]        = acc0[r] + bb;
        G[(long)(16 + mrow) * CG + col] = acc1[r] + bb;
    }
}

// Elementwise LSTM cell update. Writes h (f16, for next-step WMMA) and
// optionally appends h into the decoder hidden-state buffer.
__global__ void lstm_point_kernel(const float* __restrict__ G,
                                  float* __restrict__ c,
                                  _Float16* __restrict__ h16,
                                  _Float16* __restrict__ hs_out) {
    int i = blockIdx.x * blockDim.x + threadIdx.x;
    if (i >= CB * CH) return;
    int b = i >> 9, j = i & (CH - 1);
    const float* g = G + (long)b * CG;
    float gi = g[j], gf = g[CH + j], gg = g[2 * CH + j], go = g[3 * CH + j];
    float cn = sigf(gf) * c[i] + sigf(gi) * tanhf(gg);
    c[i] = cn;
    float hn = sigf(go) * tanhf(cn);
    h16[i] = (_Float16)hn;
    if (hs_out) hs_out[i] = (_Float16)hn;
}

// Generic small GEMM: C(M x N, f32) = A(M x K, f16) @ W(N x K, f16)^T + bias.
__global__ __launch_bounds__(256)
void gemm_bias_kernel(const _Float16* __restrict__ A, int lda,
                      const _Float16* __restrict__ W,
                      const float* __restrict__ bias,
                      float* __restrict__ C, int ldc,
                      int Mtiles, int Ntiles, int K) {
    int wid = blockIdx.x * (blockDim.x >> 5) + (threadIdx.x >> 5);
    if (wid >= Mtiles * Ntiles) return;
    int lane = threadIdx.x & 31;
    int row  = lane & 15;
    int half = lane >> 4;
    int mtile = wid / Ntiles;
    int ntile = wid % Ntiles;

    v8f acc = {};
    const _Float16* arow = A + (long)(mtile * 16 + row) * lda + half * 8;
    const _Float16* brow = W + (long)(ntile * 16 + row) * K   + half * 8;
    for (int k0 = 0; k0 < K; k0 += 32) {
        v16h a = load_frag16(arow + k0);
        v16h b = load_frag16(brow + k0);
        acc = wmma_f16(a, b, acc);
    }

    int   col = ntile * 16 + (lane & 15);
    float bb  = bias[col];
#pragma unroll
    for (int r = 0; r < 8; ++r) {
        int mrow = mtile * 16 + r + half * 8;
        C[(long)mrow * ldc + col] = acc[r] + bb;
    }
}

// Final projection: logits(3040 x 30000) = HS(3040 x 512) @ fcW(30000 x 512)^T
// + fc_b, scattered into out[b, 1+t', :].  HS row index = t'*32 + b.
// Register blocking: each wave computes a 2(M) x 5(N) grid of 16x16 tiles:
// 10 f32 accumulators (80 VGPRs), 2 a-frags + 5 b-frags per K-step -> 10 WMMAs
// per 7 fragment loads (~11 KB of L2 traffic per output tile vs 32 KB naive).
#define FC_MG 95    // 190 M-tiles / 2
#define FC_NG 375   // 1875 N-tiles / 5
__global__ __launch_bounds__(256)
void fc_kernel(const _Float16* __restrict__ HS,
               const _Float16* __restrict__ W,
               const float* __restrict__ bias,
               float* __restrict__ out) {
    int wid = blockIdx.x * (blockDim.x >> 5) + (threadIdx.x >> 5);
    if (wid >= FC_MG * FC_NG) return;
    int lane = threadIdx.x & 31;
    int row  = lane & 15;
    int half = lane >> 4;
    // consecutive waves share the same N-group -> W tile reuse in cache
    int mg = wid % FC_MG;
    int ng = wid / FC_MG;
    int mtile0 = mg * 2;
    int ntile0 = ng * 5;

    v8f acc[2][5];
#pragma unroll
    for (int m = 0; m < 2; ++m)
#pragma unroll
        for (int n = 0; n < 5; ++n) acc[m][n] = v8f{};

    const _Float16* a0 = HS + (long)(mtile0 * 16 + row)      * CH + half * 8;
    const _Float16* a1 = HS + (long)(mtile0 * 16 + 16 + row) * CH + half * 8;
    const _Float16* bw[5];
#pragma unroll
    for (int n = 0; n < 5; ++n)
        bw[n] = W + (long)((ntile0 + n) * 16 + row) * CH + half * 8;

#pragma unroll 2
    for (int k0 = 0; k0 < CH; k0 += 32) {
        v16h af0 = load_frag16(a0 + k0);
        v16h af1 = load_frag16(a1 + k0);
#pragma unroll
        for (int n = 0; n < 5; ++n) {
            v16h bf = load_frag16(bw[n] + k0);
            acc[0][n] = wmma_f16(af0, bf, acc[0][n]);
            acc[1][n] = wmma_f16(af1, bf, acc[1][n]);
        }
    }

#pragma unroll
    for (int n = 0; n < 5; ++n) {
        int   col = (ntile0 + n) * 16 + (lane & 15);
        float bb  = bias[col];
#pragma unroll
        for (int m = 0; m < 2; ++m) {
#pragma unroll
            for (int r = 0; r < 8; ++r) {
                int mrow = (mtile0 + m) * 16 + r + half * 8;  // = t'*32 + b
                int tp = mrow >> 5;
                int b  = mrow & 31;
                out[(long)b * CT * CV + (long)(tp + 1) * CV + col] =
                    acc[m][n][r] + bb;
            }
        }
    }
}

// ---------------------------------------------------------------------------
// Host launcher
// ---------------------------------------------------------------------------

extern "C" void kernel_launch(void* const* d_in, const int* in_sizes, int n_in,
                              void* d_out, int out_size, void* d_ws, size_t ws_size,
                              hipStream_t stream) {
    (void)in_sizes; (void)n_in; (void)out_size; (void)ws_size;

    const int*   Problem  = (const int*)  d_in[0];
    const int*   Formula  = (const int*)  d_in[1];
    const float* enc_emb  = (const float*)d_in[2];
    const float* dec_emb  = (const float*)d_in[3];
    const float* Wih_f    = (const float*)d_in[4];
    const float* Whh_f    = (const float*)d_in[5];
    const float* b_f      = (const float*)d_in[6];
    const float* Wih_b    = (const float*)d_in[7];
    const float* Whh_b    = (const float*)d_in[8];
    const float* b_b      = (const float*)d_in[9];
    const float* Wh       = (const float*)d_in[10];
    const float* bh       = (const float*)d_in[11];
    const float* Wc       = (const float*)d_in[12];
    const float* bc       = (const float*)d_in[13];
    const float* Wih_d    = (const float*)d_in[14];
    const float* Whh_d    = (const float*)d_in[15];
    const float* b_d      = (const float*)d_in[16];
    const float* fcW      = (const float*)d_in[17];
    const float* fcb      = (const float*)d_in[18];
    float* out = (float*)d_out;

    // ---- workspace layout (~52 MB) ----
    char* w = (char*)d_ws;
    size_t off = 0;
    auto alloc = [&](size_t bytes) -> char* {
        char* p = w + off;
        off += (bytes + 255) & ~(size_t)255;
        return p;
    };
    _Float16* Xf     = (_Float16*)alloc((size_t)CB * CS   * CKP * 2);
    _Float16* Xb     = (_Float16*)alloc((size_t)CB * CS   * CKP * 2);
    _Float16* Xd     = (_Float16*)alloc((size_t)CB * CTm1 * CKP * 2);
    _Float16* Wih_f16[3];
    Wih_f16[0] = (_Float16*)alloc((size_t)CG * CKP * 2);
    Wih_f16[1] = (_Float16*)alloc((size_t)CG * CKP * 2);
    Wih_f16[2] = (_Float16*)alloc((size_t)CG * CKP * 2);
    _Float16* Whh_f16[3];
    Whh_f16[0] = (_Float16*)alloc((size_t)CG * CH * 2);
    Whh_f16[1] = (_Float16*)alloc((size_t)CG * CH * 2);
    Whh_f16[2] = (_Float16*)alloc((size_t)CG * CH * 2);
    _Float16* Wh16   = (_Float16*)alloc((size_t)CH * 2 * CH * 2);
    _Float16* Wc16   = (_Float16*)alloc((size_t)CH * 2 * CH * 2);
    _Float16* fcW16  = (_Float16*)alloc((size_t)CV * CH * 2);
    float*    Gf     = (float*)   alloc((size_t)CB * CG * 4);
    float*    Gb     = (float*)   alloc((size_t)CB * CG * 4);
    float*    Gd     = (float*)   alloc((size_t)CB * CG * 4);
    float*    cf     = (float*)   alloc((size_t)CB * CH * 4);
    float*    cb_    = (float*)   alloc((size_t)CB * CH * 4);
    float*    cd     = (float*)   alloc((size_t)CB * CH * 4);
    _Float16* hf16   = (_Float16*)alloc((size_t)CB * CH * 2);
    _Float16* hb16   = (_Float16*)alloc((size_t)CB * CH * 2);
    _Float16* hd16   = (_Float16*)alloc((size_t)CB * CH * 2);
    _Float16* hcat   = (_Float16*)alloc((size_t)CB * 2 * CH * 2);
    _Float16* ccat   = (_Float16*)alloc((size_t)CB * 2 * CH * 2);
    float*    h0f32  = (float*)   alloc((size_t)CB * CH * 4);
    _Float16* HS16   = (_Float16*)alloc((size_t)CTm1 * CB * CH * 2);

    const int TPB = 256;
    auto blocks = [](long n, int t) { return (unsigned)((n + t - 1) / t); };

    // ---- embeddings + weight conversion ----
    enc_gather_kernel<<<blocks((long)CB * CS * CKP, TPB), TPB, 0, stream>>>(
        Problem, enc_emb, Xf, Xb);
    dec_gather_kernel<<<blocks((long)CB * CTm1 * CKP, TPB), TPB, 0, stream>>>(
        Formula, dec_emb, Xd);

    cvt_pad_kernel<<<blocks((long)CG * CKP, TPB), TPB, 0, stream>>>(Wih_f, Wih_f16[0], CG, CE, CKP);
    cvt_pad_kernel<<<blocks((long)CG * CKP, TPB), TPB, 0, stream>>>(Wih_b, Wih_f16[1], CG, CE, CKP);
    cvt_pad_kernel<<<blocks((long)CG * CKP, TPB), TPB, 0, stream>>>(Wih_d, Wih_f16[2], CG, CE, CKP);
    cvt_pad_kernel<<<blocks((long)CG * CH,  TPB), TPB, 0, stream>>>(Whh_f, Whh_f16[0], CG, CH, CH);
    cvt_pad_kernel<<<blocks((long)CG * CH,  TPB), TPB, 0, stream>>>(Whh_b, Whh_f16[1], CG, CH, CH);
    cvt_pad_kernel<<<blocks((long)CG * CH,  TPB), TPB, 0, stream>>>(Whh_d, Whh_f16[2], CG, CH, CH);
    cvt_pad_kernel<<<blocks((long)CH * 2 * CH, TPB), TPB, 0, stream>>>(Wh, Wh16, CH, 2 * CH, 2 * CH);
    cvt_pad_kernel<<<blocks((long)CH * 2 * CH, TPB), TPB, 0, stream>>>(Wc, Wc16, CH, 2 * CH, 2 * CH);
    cvt_pad_kernel<<<blocks((long)CV * CH,  TPB), TPB, 0, stream>>>(fcW, fcW16, CV, CH, CH);

    // ---- zero-init recurrent state ----
    zero_u32_kernel<<<blocks(CB * CH, TPB), TPB, 0, stream>>>((unsigned*)cf,  CB * CH);
    zero_u32_kernel<<<blocks(CB * CH, TPB), TPB, 0, stream>>>((unsigned*)cb_, CB * CH);
    zero_u32_kernel<<<blocks(CB * CH / 2, TPB), TPB, 0, stream>>>((unsigned*)hf16, CB * CH / 2);
    zero_u32_kernel<<<blocks(CB * CH / 2, TPB), TPB, 0, stream>>>((unsigned*)hb16, CB * CH / 2);

    // ---- encoder: forward direction ----
    for (int t = 0; t < CS; ++t) {
        lstm_gates_kernel<<<16, 256, 0, stream>>>(
            Xf + (long)t * CKP, CS * CKP, Wih_f16[0], hf16, Whh_f16[0], b_f, Gf);
        lstm_point_kernel<<<blocks(CB * CH, TPB), TPB, 0, stream>>>(Gf, cf, hf16, nullptr);
    }
    // ---- encoder: backward direction (Xb already time-reversed) ----
    for (int t = 0; t < CS; ++t) {
        lstm_gates_kernel<<<16, 256, 0, stream>>>(
            Xb + (long)t * CKP, CS * CKP, Wih_f16[1], hb16, Whh_f16[1], b_b, Gb);
        lstm_point_kernel<<<blocks(CB * CH, TPB), TPB, 0, stream>>>(Gb, cb_, hb16, nullptr);
    }

    // ---- h0 / c0 bridge: [h_f|h_b] @ Wh^T + bh, [c_f|c_b] @ Wc^T + bc ----
    concat_hc_kernel<<<blocks(CB * 2 * CH, TPB), TPB, 0, stream>>>(hf16, hb16, cf, cb_, hcat, ccat);
    gemm_bias_kernel<<<8, 256, 0, stream>>>(hcat, 2 * CH, Wh16, bh, h0f32, CH, 2, CH / 16, 2 * CH);
    gemm_bias_kernel<<<8, 256, 0, stream>>>(ccat, 2 * CH, Wc16, bc, cd,    CH, 2, CH / 16, 2 * CH);
    cvt_pad_kernel<<<blocks(CB * CH, TPB), TPB, 0, stream>>>(h0f32, hd16, CB, CH, CH);

    // ---- decoder ----
    for (int t = 0; t < CTm1; ++t) {
        lstm_gates_kernel<<<16, 256, 0, stream>>>(
            Xd + (long)t * CKP, CTm1 * CKP, Wih_f16[2], hd16, Whh_f16[2], b_d, Gd);
        lstm_point_kernel<<<blocks(CB * CH, TPB), TPB, 0, stream>>>(
            Gd, cd, hd16, HS16 + (long)t * CB * CH);
    }

    // ---- output projection + t=0 zero slab ----
    zero_t0_kernel<<<blocks((long)CB * CV, TPB), TPB, 0, stream>>>(out);
    {
        const long waves = (long)FC_MG * FC_NG;   // 95 * 375 = 35625
        fc_kernel<<<blocks(waves, 8), 256, 0, stream>>>(HS16, fcW16, fcb, out);
    }
}
